// NTXentLoss_5961414607650
// MI455X (gfx1250) — compile-verified, matrix operand order checked by reading, not run on hardware
//
#include <hip/hip_runtime.h>
#include <stdint.h>

// NT-Xent loss for MI455X (gfx1250), compute-bound Gram matrix via bf16 WMMA
// with hi/lo fp32-recovery split + flash-style online logsumexp.

typedef __attribute__((ext_vector_type(16))) __bf16 v16bf;
typedef __attribute__((ext_vector_type(8)))  float  v8f;

union Frag { v16bf v; uint4 q[2]; };

constexpr int   D_DIM  = 128;
constexpr int   KCHUNK = 32;
constexpr int   NCH    = D_DIM / KCHUNK;   // 4
constexpr float TEMP   = 0.07f;
constexpr float LN2    = 0.69314718055994530942f;

// ---- prep: split fp32 z (concat of z_i,z_j) into bf16 hi + bf16 residual ----
__global__ void ntxent_prep(const float* __restrict__ zi, const float* __restrict__ zj,
                            unsigned short* __restrict__ zhi, unsigned short* __restrict__ zlo,
                            int halfElems, int total) {
  int idx = blockIdx.x * blockDim.x + threadIdx.x;
  if (idx >= total) return;
  float v = (idx < halfElems) ? zi[idx] : zj[idx - halfElems];
  unsigned int b  = __float_as_uint(v);
  unsigned int r  = b + 0x7fffu + ((b >> 16) & 1u);        // RNE to bf16
  unsigned short h = (unsigned short)(r >> 16);
  float hf  = __uint_as_float(((unsigned int)h) << 16);
  float res = v - hf;
  unsigned int b2 = __float_as_uint(res);
  unsigned int r2 = b2 + 0x7fffu + ((b2 >> 16) & 1u);
  zhi[idx] = h;
  zlo[idx] = (unsigned short)(r2 >> 16);
}

__global__ void ntxent_zero(float* out) { out[0] = 0.0f; }

// ---- main: each wave owns 16 rows, sweeps all column tiles with WMMA ----
__global__ __launch_bounds__(256) void ntxent_main(const unsigned short* __restrict__ zhiu,
                                                   const unsigned short* __restrict__ zlou,
                                                   float* __restrict__ out, int Brows) {
  const int N    = 2 * Brows;
  const int lane = threadIdx.x & 31;
  const int wave = threadIdx.x >> 5;
  const int lrow = lane & 15;
  const int g    = lane >> 4;                 // lane half-group
  const int m0   = (blockIdx.x * 8 + wave) * 16;

  // Resident A fragments (hi/lo) for rows m0..m0+15.
  // 16-bit A/B layout: element e <-> k = e + 8*(e>=8) + 8*g  (contiguous runs of 8)
  Frag ahi[NCH], alo[NCH];
  {
    const unsigned short* ra = zhiu + (size_t)(m0 + lrow) * D_DIM;
    const unsigned short* rb = zlou + (size_t)(m0 + lrow) * D_DIM;
#pragma unroll
    for (int c = 0; c < NCH; ++c) {
      const int k0 = c * KCHUNK + 8 * g;
      ahi[c].q[0] = *(const uint4*)(ra + k0);
      ahi[c].q[1] = *(const uint4*)(ra + k0 + 16);
      alo[c].q[0] = *(const uint4*)(rb + k0);
      alo[c].q[1] = *(const uint4*)(rb + k0 + 16);
    }
  }

  const float SC     = 1.0f / (TEMP * LN2);            // base-2 domain
  const int   diagR  = (lane < 16) ? lrow : (lane - 24);
  const bool  hasDiag = (lane < 8) || (lane >= 24);
  const int   posT   = (m0 < Brows) ? (m0 + Brows) : (m0 - Brows);

  float mx[8], sm[8];
#pragma unroll
  for (int r = 0; r < 8; ++r) { mx[r] = -1.0e30f; sm[r] = 0.0f; }
  float posv = -1.0e30f;

  for (int j0 = 0; j0 < N; j0 += 16) {
    const unsigned short* rh = zhiu + (size_t)(j0 + lrow) * D_DIM;
    const unsigned short* rl = zlou + (size_t)(j0 + lrow) * D_DIM;
    Frag bhi[NCH], blo[NCH];
#pragma unroll
    for (int c = 0; c < NCH; ++c) {
      const int k0 = c * KCHUNK + 8 * g;
      bhi[c].q[0] = *(const uint4*)(rh + k0);
      bhi[c].q[1] = *(const uint4*)(rh + k0 + 16);
      blo[c].q[0] = *(const uint4*)(rl + k0);
      blo[c].q[1] = *(const uint4*)(rl + k0 + 16);
    }
    if (j0 + 16 < N) {                       // pull next panel toward WGP$
      __builtin_prefetch(rh + (size_t)16 * D_DIM, 0, 0);
      __builtin_prefetch(rl + (size_t)16 * D_DIM, 0, 0);
    }

    // dot ~= hi*hi + hi*lo + lo*hi  (two acc chains to shorten RAW depth)
    v8f acc0 = {}, acc1 = {};
#pragma unroll
    for (int c = 0; c < NCH; ++c) {
      acc0 = __builtin_amdgcn_wmma_f32_16x16x32_bf16(false, ahi[c].v, false, bhi[c].v,
                                                     (short)0, acc0, false, false);
      acc1 = __builtin_amdgcn_wmma_f32_16x16x32_bf16(false, ahi[c].v, false, blo[c].v,
                                                     (short)0, acc1, false, false);
      acc1 = __builtin_amdgcn_wmma_f32_16x16x32_bf16(false, alo[c].v, false, bhi[c].v,
                                                     (short)0, acc1, false, false);
    }

    const bool dTile = (j0 == m0);
    const bool pTile = (j0 == posT);
#pragma unroll
    for (int r = 0; r < 8; ++r) {
      float v = (acc0[r] + acc1[r]) * SC;
      if (hasDiag && r == diagR) {
        if (pTile) posv = v;                 // positive pair (kept in sum too)
        if (dTile) v = -3.0e38f;             // mask self-similarity
      }
      float nm = fmaxf(mx[r], v);            // lane-local online LSE (base 2)
      sm[r] = sm[r] * exp2f(mx[r] - nm) + exp2f(v - nm);
      mx[r] = nm;
    }
  }

  // Merge 16 lane-partials per row; add the double-counted positive term.
  float rowacc = 0.0f;
#pragma unroll
  for (int r = 0; r < 8; ++r) {
    float M = mx[r], S = sm[r];
#pragma unroll
    for (int off = 1; off < 16; off <<= 1) {
      float Mo = __shfl_xor(M, off, 32);
      float So = __shfl_xor(S, off, 32);
      float nm = fmaxf(M, Mo);
      S = S * exp2f(M - nm) + So * exp2f(Mo - nm);
      M = nm;
    }
    float p = (hasDiag && diagR == r) ? posv : -1.0e30f;
#pragma unroll
    for (int off = 1; off < 16; off <<= 1)
      p = fmaxf(p, __shfl_xor(p, off, 32));
    S += exp2f(p - M);                       // aug row: pos counted twice
    rowacc += M + log2f(S) - p;              // (lse - pos) in base-2
  }

  float val = (lrow == 0) ? rowacc : 0.0f;   // lanes 0/16 hold rows 0-7 / 8-15
#pragma unroll
  for (int off = 1; off < 32; off <<= 1) val += __shfl_xor(val, off, 32);
  if (lane == 0) atomicAdd(out, val * (LN2 / (float)N));
}

extern "C" void kernel_launch(void* const* d_in, const int* in_sizes, int n_in,
                              void* d_out, int out_size, void* d_ws, size_t ws_size,
                              hipStream_t stream) {
  const float* zi = (const float*)d_in[0];
  const float* zj = (const float*)d_in[1];
  const int BD    = in_sizes[0];        // B * 128
  const int Brows = BD / D_DIM;         // 4096
  const int N     = 2 * Brows;          // 8192
  const int total = N * D_DIM;

  unsigned short* zhi = (unsigned short*)d_ws;
  unsigned short* zlo = zhi + (size_t)total;   // 2 MB + 2 MB in d_ws

  ntxent_zero<<<1, 1, 0, stream>>>((float*)d_out);
  ntxent_prep<<<(total + 255) / 256, 256, 0, stream>>>(zi, zj, zhi, zlo, BD, total);
  ntxent_main<<<N / 128, 256, 0, stream>>>(zhi, zlo, (float*)d_out, Brows);
}
